// Deformable_MatMul3_26147760898634
// MI455X (gfx1250) — matrix operand order repaired and checked by reading, not compiled, not run on hardware
//
#include <hip/hip_runtime.h>

typedef __attribute__((ext_vector_type(16))) _Float16 v16h;
typedef __attribute__((ext_vector_type(8)))  _Float16 v8h;
typedef __attribute__((ext_vector_type(8)))  float    v8f;

#define C_IN   256
#define HW     4096
#define O_CH   256
#define NREG   8
#define PTILE  64          // pixels per workgroup
#define OTILE  128         // output channels per workgroup
#define KCHUNK 64          // K staged per barrier round (2 WMMA k-steps)
#define XS_STRIDE 264      // halfs per pixel row in LDS (256 + 8 pad, 528B, 16B aligned)
#define WS_STRIDE 72       // halfs per o-row in LDS weight chunk (64 + 8 pad, 144B)

#define USE_ASYNC_LDS 1    // stage weights with global_load_async_to_lds_b128

// ---------------------------------------------------------------------------
// Prep: Wp[r][o][c] = mat1[o,c,r] * (use_alpha ? Alpha[r] : 1), converted f16.
// ---------------------------------------------------------------------------
__global__ __launch_bounds__(256) void prep_weights(
    const float* __restrict__ mat1, const float* __restrict__ alpha,
    const int* __restrict__ use_alpha, _Float16* __restrict__ wp) {
  int tid = blockIdx.x * 256 + threadIdx.x;       // [r(3) | o(8) | c(8)]
  int c = tid & (C_IN - 1);
  int o = (tid >> 8) & (O_CH - 1);
  int r = tid >> 16;
  float a = use_alpha[0] ? alpha[r] : 1.0f;
  wp[tid] = (_Float16)(mat1[(o * C_IN + c) * NREG + r] * a);
}

// ---------------------------------------------------------------------------
// Main: per WG computes out tile [OTILE x PTILE].
//   A = W'_r (M=O rows, K=C), B = mat0 tile (K=C, N=pixels), f32 accumulate.
//   8 waves arranged 4(O) x 2(pix); each wave owns 32 O x 32 pix = 2x2 WMMA
//   tiles. Regions outer; K loop in chunks of 64 (8 WMMA per barrier round);
//   mask fold after each region's GEMM.
// ---------------------------------------------------------------------------
__global__ __launch_bounds__(256) void deform_mm(
    const float* __restrict__ mat0, const _Float16* __restrict__ wp,
    const float* __restrict__ mask, float* __restrict__ out) {
  extern __shared__ __align__(16) char smem[];
  _Float16* Xs = (_Float16*)smem;                    // PTILE x XS_STRIDE
  _Float16* Wc = Xs + PTILE * XS_STRIDE;             // OTILE x WS_STRIDE
  float*    Ms = (float*)(Wc + OTILE * WS_STRIDE);   // NREG x PTILE
  const unsigned WC_BASE = (unsigned)(PTILE * XS_STRIDE) * 2u;  // Wc byte offset in LDS

  const int t     = threadIdx.x;
  const int b     = blockIdx.x >> 6;                 // 64 pixel-tiles per batch
  const int pp    = (blockIdx.x & 63) * PTILE;       // pixel base within batch
  const int obase = blockIdx.y * OTILE;

  // ---- stage mat0 tile: 64 pix x 256 c, f32 -> f16, transposed into [pix][c]
  {
    int pix = t & 63;
    int c0  = t >> 6;                                // 0..3
    const float* src = mat0 + (size_t)b * C_IN * HW + pp + pix;
    #pragma unroll 4
    for (int cb = 0; cb < C_IN; cb += 4) {
      int c = cb + c0;
      Xs[pix * XS_STRIDE + c] = (_Float16)src[(size_t)c * HW];
    }
  }
  // ---- stage mask tile: 8 r x 64 pix
  for (int it = 0; it < 2; ++it) {
    int idx = it * 256 + t;
    int r = idx >> 6, pix = idx & 63;
    Ms[idx] = mask[r * HW + pp + pix];
  }
  __syncthreads();

  const int wid    = t >> 5;
  const int lane   = t & 31;
  const int lrow   = lane & 15;
  const int lhi    = lane >> 4;
  const int wave_o = wid >> 1;                       // 0..3 -> 32 O each
  const int wave_p = wid & 1;                        // 0..1 -> 32 pix each

  v8f acc[2][2] = {};

  for (int r = 0; r < NREG; ++r) {
    v8f racc[2][2] = {};

    for (int kt0 = 0; kt0 < C_IN / KCHUNK; ++kt0) {  // 4 rounds of K=64
      __syncthreads();                               // Wc safe to overwrite

      // ---- stage weight chunk: 128 o x 64 halfs ----------------------------
      const _Float16* wsrc = wp + ((size_t)r * O_CH + obase) * C_IN + kt0 * KCHUNK;
#if USE_ASYNC_LDS
      // CDNA5 async global->LDS copy (GVS mode), tracked by ASYNCcnt.
      #pragma unroll
      for (int it = 0; it < 4; ++it) {
        int chunk = it * 256 + t;                    // 1024 x 16B transfers
        int o  = chunk >> 3;
        int c8 = (chunk & 7) * 8;
        unsigned ldsOff = WC_BASE + (unsigned)(o * WS_STRIDE + c8) * 2u;
        unsigned gOff   = (unsigned)(o * C_IN + c8) * 2u;
        asm volatile("global_load_async_to_lds_b128 %0, %1, %2"
                     :
                     : "v"(ldsOff), "v"(gOff), "s"(wsrc)
                     : "memory");
      }
      asm volatile("s_wait_asynccnt 0" ::: "memory");
#else
      #pragma unroll
      for (int it = 0; it < 4; ++it) {
        int chunk = it * 256 + t;
        int o  = chunk >> 3;
        int c8 = (chunk & 7) * 8;
        v8h v = *(const v8h*)(wsrc + (size_t)o * C_IN + c8);
        *(v8h*)(Wc + o * WS_STRIDE + c8) = v;
      }
#endif
      __syncthreads();

      // ---- two K=32 sub-steps per staged chunk: 8 WMMAs per barrier round --
      #pragma unroll
      for (int ks = 0; ks < 2; ++ks) {
        // A fragments (weights): lane row = lrow, K halves per ISA layout
        v16h a[2];
        #pragma unroll
        for (int mo = 0; mo < 2; ++mo) {
          int row = wave_o * 32 + mo * 16 + lrow;
          const _Float16* p = Wc + row * WS_STRIDE + ks * 32 + lhi * 8;
          v8h lo = *(const v8h*)p;                   // K = koff + 0..7
          v8h hi = *(const v8h*)(p + 16);            // K = koff + 16..23
          #pragma unroll
          for (int i = 0; i < 8; ++i) { a[mo][i] = lo[i]; a[mo][i + 8] = hi[i]; }
        }
        // B fragments (pixels): lane col = lrow, K halves per ISA layout
        v16h bf[2];
        #pragma unroll
        for (int pt = 0; pt < 2; ++pt) {
          int pixr = wave_p * 32 + pt * 16 + lrow;
          const _Float16* p = Xs + pixr * XS_STRIDE + kt0 * KCHUNK + ks * 32 + lhi * 8;
          v8h lo = *(const v8h*)p;
          v8h hi = *(const v8h*)(p + 16);
          #pragma unroll
          for (int i = 0; i < 8; ++i) { bf[pt][i] = lo[i]; bf[pt][i + 8] = hi[i]; }
        }
        #pragma unroll
        for (int mo = 0; mo < 2; ++mo)
          #pragma unroll
          for (int pt = 0; pt < 2; ++pt)
            racc[mo][pt] = __builtin_amdgcn_wmma_f32_16x16x32_f16(
                false, a[mo], false, bf[pt], (short)0, racc[mo][pt], false, false);
      }
    }

    // ---- fold region into output accumulator, weighted by mask[r, pix] ----
    #pragma unroll
    for (int pt = 0; pt < 2; ++pt) {
      float s = Ms[r * PTILE + wave_p * 32 + pt * 16 + lrow];
      #pragma unroll
      for (int mo = 0; mo < 2; ++mo)
        #pragma unroll
        for (int i = 0; i < 8; ++i)
          acc[mo][pt][i] += s * racc[mo][pt][i];
    }
  }

  // ---- store: D layout — lane holds col N=lrow, VGPR v holds row M=v+8*lhi
  #pragma unroll
  for (int mo = 0; mo < 2; ++mo)
    #pragma unroll
    for (int pt = 0; pt < 2; ++pt) {
      int pixg = pp + wave_p * 32 + pt * 16 + lrow;
      #pragma unroll
      for (int v = 0; v < 8; ++v) {
        int o = obase + wave_o * 32 + mo * 16 + v + lhi * 8;
        out[((size_t)(b * O_CH + o)) * HW + pixg] = acc[mo][pt][v];
      }
    }
}

extern "C" void kernel_launch(void* const* d_in, const int* in_sizes, int n_in,
                              void* d_out, int out_size, void* d_ws, size_t ws_size,
                              hipStream_t stream) {
  const float* mat0      = (const float*)d_in[0];
  const float* mat1      = (const float*)d_in[1];
  const float* mask      = (const float*)d_in[2];
  const float* alpha     = (const float*)d_in[3];
  const int*   use_alpha = (const int*)d_in[4];
  // d_in[5] = beta: unused by the reference computation.
  _Float16* wp = (_Float16*)d_ws;                    // NREG*O_CH*C_IN f16 = 1 MB

  prep_weights<<<(NREG * O_CH * C_IN) / 256, 256, 0, stream>>>(
      mat1, alpha, use_alpha, wp);

  size_t lds = (size_t)(PTILE * XS_STRIDE + OTILE * WS_STRIDE) * sizeof(unsigned short)
             + (size_t)NREG * PTILE * sizeof(float);  // 54272 B
  dim3 grid(512, O_CH / OTILE);                       // 512 pixel tiles x 2 O tiles
  deform_mm<<<grid, 256, lds, stream>>>(mat0, wp, mask, (float*)d_out);
}
